// GCN_graph_classification_69277822484502
// MI455X (gfx1250) — matrix operand order repaired and checked by reading, not compiled, not run on hardware
//
#include <hip/hip_runtime.h>
#include <math.h>

#define NN 8192
#define DD 128
#define HH 128
#define CC 10
#define GG 64

typedef __attribute__((ext_vector_type(16))) __bf16 v16bf;
typedef __attribute__((ext_vector_type(8)))  __bf16 v8bf;
typedef __attribute__((ext_vector_type(8)))  float  v8f;

static __device__ __forceinline__ unsigned short bf16_rne_bits(float f) {
  unsigned int u = __builtin_bit_cast(unsigned int, f);
  unsigned int r = u + 0x7FFFu + ((u >> 16) & 1u);
  return (unsigned short)(r >> 16);
}
static __device__ __forceinline__ float bf16_bits_f32(unsigned short s) {
  unsigned int u = ((unsigned int)s) << 16;
  return __builtin_bit_cast(float, u);
}
static __device__ __forceinline__ __bf16 bits_bf16(unsigned short s) {
  return __builtin_bit_cast(__bf16, s);
}

// CDNA5 async memory->LDS copy (16B per lane), tracked by ASYNCcnt.
// lds = wave-relative LDS byte address (low 32 bits of generic pointer).
static __device__ __forceinline__ void async_b128(const void* gaddr, unsigned lds) {
  asm volatile("global_load_async_to_lds_b128 %0, %1, off"
               :: "v"(lds), "v"(gaddr) : "memory");
}
#define S_WAIT_ASYNCCNT(n) asm volatile("s_wait_asynccnt " #n ::: "memory")

static __device__ __forceinline__ unsigned lds_addr(const void* p) {
  return (unsigned)(size_t)p;  // LDS aperture: addr[31:0] is the LDS offset
}

// ---------------------------------------------------------------------------
// 1) diags[i] = sum_j adj[i][j] + d1   (temp = adj + d1*I row sums)
// ---------------------------------------------------------------------------
__global__ __launch_bounds__(256) void k_rowsum(const float* __restrict__ adj,
                                                const float* __restrict__ d1p,
                                                float* __restrict__ diags) {
  __shared__ float red[256];
  const int row = blockIdx.x;
  const int t = threadIdx.x;
  const float* r = adj + (size_t)row * NN;
  float s = 0.f;
  for (int j = t; j < NN; j += 256) s += r[j];
  red[t] = s;
  __syncthreads();
  for (int off = 128; off > 0; off >>= 1) {
    if (t < off) red[t] += red[t + off];
    __syncthreads();
  }
  if (t == 0) diags[row] = red[0] + d1p[0];
}

// ---------------------------------------------------------------------------
// 2) per-row PGSO scale factors
// ---------------------------------------------------------------------------
__global__ __launch_bounds__(256) void k_scales(const float* __restrict__ diags,
                                                const float* __restrict__ pp,
                                                const float* __restrict__ qp,
                                                const float* __restrict__ rp,
                                                const float* __restrict__ c1p,
                                                const float* __restrict__ c2p,
                                                const float* __restrict__ c3p,
                                                float* __restrict__ rowscale,
                                                float* __restrict__ diagscale,
                                                float* __restrict__ drv) {
  const int i = blockIdx.x * 256 + threadIdx.x;
  const float dg = diags[i];
  rowscale[i]  = -c2p[0] * powf(dg, qp[0]);
  diagscale[i] = c3p[0] * powf(dg, pp[0]) + c1p[0];
  drv[i]       = powf(dg, rp[0]);
}

// ---------------------------------------------------------------------------
// 3) Z = Hin @ W   (8192x128 @ 128x128, tiny vs the big matmul)
// ---------------------------------------------------------------------------
__global__ __launch_bounds__(256) void k_gemm(const float* __restrict__ Hin,
                                              const float* __restrict__ W,
                                              float* __restrict__ Z) {
  const int gid = blockIdx.x * 256 + threadIdx.x;
  const int i = gid >> 7;
  const int n = gid & 127;
  const float* hr = Hin + (size_t)i * DD;
  float s = 0.f;
#pragma unroll 8
  for (int d = 0; d < DD; ++d) s = fmaf(hr[d], W[d * HH + n], s);
  Z[gid] = s;
}

// ---------------------------------------------------------------------------
// 4) ZsT_hi/lo[n][k] = split_bf16( drv[k] * Z[k][n] )   (transposed, K-contig)
// ---------------------------------------------------------------------------
__global__ void k_split_t(const float* __restrict__ Z,
                          const float* __restrict__ drv,
                          unsigned short* __restrict__ btHi,
                          unsigned short* __restrict__ btLo) {
  __shared__ float tile[32][33];
  const int k0 = blockIdx.x * 32;
  const int n0 = blockIdx.y * 32;
  const int tx = threadIdx.x;
  const int ty = threadIdx.y;
#pragma unroll
  for (int i = 0; i < 4; ++i) {
    const int kk = ty + i * 8;
    tile[kk][tx] = Z[(size_t)(k0 + kk) * HH + (n0 + tx)];
  }
  __syncthreads();
  const float dv = drv[k0 + tx];
#pragma unroll
  for (int i = 0; i < 4; ++i) {
    const int nn = ty + i * 8;
    const float v = tile[tx][nn] * dv;
    const unsigned short hb = bf16_rne_bits(v);
    const unsigned short lb = bf16_rne_bits(v - bf16_bits_f32(hb));
    const size_t o = (size_t)(n0 + nn) * NN + (k0 + tx);
    btHi[o] = hb;
    btLo[o] = lb;
  }
}

// ---------------------------------------------------------------------------
// 5) Flagship: Y = adj @ Zs (split-bf16, 3 WMMA passes), fused PGSO epilogue.
//    B tiles double-buffered in LDS via global_load_async_to_lds_b128;
//    A operand software-pipelined in registers.
//    Block: 128 threads (4 waves), 64 rows x 128 cols of output.
// ---------------------------------------------------------------------------
#define BPITCH 80   // bytes per 32-k bf16 row in LDS (64B data + pad, 16B aligned)
#define KITERS (NN / 32)

__global__ __launch_bounds__(128) void k_spmm(const float* __restrict__ adj,
                                              const unsigned short* __restrict__ btHi,
                                              const unsigned short* __restrict__ btLo,
                                              const float* __restrict__ Z,
                                              const float* __restrict__ drv,
                                              const float* __restrict__ rowscale,
                                              const float* __restrict__ diagscale,
                                              const float* __restrict__ d1p,
                                              const float* __restrict__ bias,
                                              float* __restrict__ Hout) {
  // [buffer][hi/lo][128 rows * BPITCH]
  __shared__ __align__(16) unsigned char sB[2][2][128 * BPITCH];

  const int tid   = threadIdx.x;
  const int lane  = tid & 31;
  const int wv    = tid >> 5;
  const int rbase = blockIdx.x * 64 + wv * 16;
  const int arow  = rbase + (lane & 15);
  const int ksel  = (lane >> 4) & 1;  // which K half this lane owns

  // async-stage one 128x32 bf16 tile pair (hi+lo) into buffer `buf`
  auto prefetchB = [&](int buf, int kb) {
    const size_t src = (size_t)tid * NN + kb;
#pragma unroll
    for (int c = 0; c < 4; ++c) {
      async_b128(btHi + src + c * 8, lds_addr(&sB[buf][0][tid * BPITCH + c * 16]));
      async_b128(btLo + src + c * 8, lds_addr(&sB[buf][1][tid * BPITCH + c * 16]));
    }
  };

  v8f acc[8];
#pragma unroll
  for (int j = 0; j < 8; ++j)
#pragma unroll
    for (int e = 0; e < 8; ++e) acc[j][e] = 0.0f;

  // prologue: stage B tile 0, load A tile 0 into registers
  prefetchB(0, 0);
  const size_t arowoff = (size_t)arow * NN;
  float4 A0, A1, A2, A3;
  {
    const float4* p0 = (const float4*)(adj + arowoff + ksel * 8);
    const float4* p1 = (const float4*)(adj + arowoff + ksel * 8 + 16);
    A0 = p0[0]; A1 = p0[1]; A2 = p1[0]; A3 = p1[1];
  }

  for (int it = 0; it < KITERS; ++it) {
    const int cur = it & 1;
    const bool more = (it + 1) < KITERS;

    __syncthreads();  // all waves finished reading buffer cur^1 (prev compute)
    if (more) {
      prefetchB(cur ^ 1, (it + 1) * 32);
      S_WAIT_ASYNCCNT(8);  // in-order: oldest 8 (current buffer) have landed
    } else {
      S_WAIT_ASYNCCNT(0);
    }
    __syncthreads();  // everyone's rows of buffer `cur` visible

    // issue next A-tile loads early (hidden behind this iteration's WMMAs)
    const int kbn = more ? (it + 1) * 32 : 0;  // clamp keeps last read in-bounds
    const float4* np0 = (const float4*)(adj + arowoff + kbn + ksel * 8);
    const float4* np1 = (const float4*)(adj + arowoff + kbn + ksel * 8 + 16);
    const float4 N0 = np0[0], N1 = np0[1], N2 = np1[0], N3 = np1[1];

    // split current A tile (16x32 f32) into bf16 hi/lo fragments
    const float fr[16] = {A0.x, A0.y, A0.z, A0.w, A1.x, A1.y, A1.z, A1.w,
                          A2.x, A2.y, A2.z, A2.w, A3.x, A3.y, A3.z, A3.w};
    v16bf ahi, alo;
#pragma unroll
    for (int e = 0; e < 16; ++e) {
      const unsigned short hb = bf16_rne_bits(fr[e]);
      const unsigned short lb = bf16_rne_bits(fr[e] - bf16_bits_f32(hb));
      ahi[e] = bits_bf16(hb);
      alo[e] = bits_bf16(lb);
    }

#pragma unroll
    for (int j = 0; j < 8; ++j) {
      const int n = j * 16 + (lane & 15);
      const unsigned char* ph = &sB[cur][0][n * BPITCH + ksel * 16];
      const unsigned char* pl = &sB[cur][1][n * BPITCH + ksel * 16];
      const v8bf h0 = *(const v8bf*)(ph);
      const v8bf h1 = *(const v8bf*)(ph + 32);
      const v8bf l0 = *(const v8bf*)(pl);
      const v8bf l1 = *(const v8bf*)(pl + 32);
      const v16bf bh = __builtin_shufflevector(h0, h1, 0, 1, 2, 3, 4, 5, 6, 7,
                                               8, 9, 10, 11, 12, 13, 14, 15);
      const v16bf bl = __builtin_shufflevector(l0, l1, 0, 1, 2, 3, 4, 5, 6, 7,
                                               8, 9, 10, 11, 12, 13, 14, 15);
      acc[j] = __builtin_amdgcn_wmma_f32_16x16x32_bf16(false, ahi, false, bh,
                                                       (short)0, acc[j], false, false);
      acc[j] = __builtin_amdgcn_wmma_f32_16x16x32_bf16(false, ahi, false, bl,
                                                       (short)0, acc[j], false, false);
      acc[j] = __builtin_amdgcn_wmma_f32_16x16x32_bf16(false, alo, false, bh,
                                                       (short)0, acc[j], false, false);
    }

    A0 = N0; A1 = N1; A2 = N2; A3 = N3;
  }

  // fused epilogue (C layout: VGPR r -> M = r (lanes 0-15) or 8+r (16-31))
  const float d1v = d1p[0];
#pragma unroll
  for (int j = 0; j < 8; ++j) {
    const int n = j * 16 + (lane & 15);
    const float bn = bias[n];
#pragma unroll
    for (int r = 0; r < 8; ++r) {
      const int i = rbase + (ksel ? (r + 8) : r);
      const float zin = Z[(size_t)i * HH + n];
      const float zs = drv[i] * zin;
      float g = rowscale[i] * (acc[j][r] + d1v * zs) + diagscale[i] * zin + bn;
      Hout[(size_t)i * HH + n] = g > 0.0f ? g : 0.0f;
    }
  }
}

// ---------------------------------------------------------------------------
// 6) segment offsets from sorted batch vector (int64)
// ---------------------------------------------------------------------------
__global__ __launch_bounds__(256) void k_offsets(const long long* __restrict__ batch,
                                                 int* __restrict__ offs) {
  __shared__ int cnt[GG];
  const int t = threadIdx.x;
  if (t < GG) cnt[t] = 0;
  __syncthreads();
  for (int i = t; i < NN; i += 256) atomicAdd(&cnt[(int)batch[i]], 1);
  __syncthreads();
  if (t == 0) {
    int run = 0;
    offs[0] = 0;
    for (int g = 0; g < GG; ++g) { run += cnt[g]; offs[g + 1] = run; }
  }
}

// ---------------------------------------------------------------------------
// 7) global_add_pool
// ---------------------------------------------------------------------------
__global__ __launch_bounds__(128) void k_pool(const float* __restrict__ H,
                                              const int* __restrict__ offs,
                                              float* __restrict__ pooled) {
  const int g = blockIdx.x;
  const int n = threadIdx.x;
  const int a = offs[g], b = offs[g + 1];
  float s = 0.f;
  for (int i = a; i < b; ++i) s += H[(size_t)i * HH + n];
  pooled[g * HH + n] = s;
}

// ---------------------------------------------------------------------------
// 8) FC head + log_softmax
// ---------------------------------------------------------------------------
__global__ __launch_bounds__(64) void k_head(const float* __restrict__ pooled,
                                             const float* __restrict__ fcw,
                                             const float* __restrict__ fcb,
                                             float* __restrict__ out) {
  const int g = threadIdx.x;  // 64 threads, one graph each
  float logit[CC];
  float m = -1e30f;
#pragma unroll
  for (int c = 0; c < CC; ++c) {
    float s = fcb[c];
    for (int n = 0; n < HH; ++n) s = fmaf(pooled[g * HH + n], fcw[n * CC + c], s);
    logit[c] = s;
    m = fmaxf(m, s);
  }
  float lse = 0.f;
#pragma unroll
  for (int c = 0; c < CC; ++c) lse += expf(logit[c] - m);
  lse = logf(lse) + m;
#pragma unroll
  for (int c = 0; c < CC; ++c) out[g * CC + c] = logit[c] - lse;
}

// ---------------------------------------------------------------------------
extern "C" void kernel_launch(void* const* d_in, const int* in_sizes, int n_in,
                              void* d_out, int out_size, void* d_ws, size_t ws_size,
                              hipStream_t stream) {
  (void)in_sizes; (void)n_in; (void)out_size; (void)ws_size;
  const float* x         = (const float*)d_in[0];
  const float* adj       = (const float*)d_in[1];
  const long long* batch = (const long long*)d_in[2];
  const float* pp  = (const float*)d_in[3];
  const float* qp  = (const float*)d_in[4];
  const float* rp  = (const float*)d_in[5];
  const float* c1p = (const float*)d_in[6];
  const float* c2p = (const float*)d_in[7];
  const float* c3p = (const float*)d_in[8];
  const float* d1p = (const float*)d_in[9];
  const float* Ws[3] = {(const float*)d_in[10], (const float*)d_in[12], (const float*)d_in[14]};
  const float* bs[3] = {(const float*)d_in[11], (const float*)d_in[13], (const float*)d_in[15]};
  const float* fcw = (const float*)d_in[16];
  const float* fcb = (const float*)d_in[17];
  float* out = (float*)d_out;

  char* ws = (char*)d_ws;
  size_t off = 0;
  auto alloc = [&](size_t bytes) -> void* {
    void* p = ws + off;
    off += (bytes + 255) & ~(size_t)255;
    return p;
  };
  float* diags     = (float*)alloc((size_t)NN * 4);
  float* rowscale  = (float*)alloc((size_t)NN * 4);
  float* diagscale = (float*)alloc((size_t)NN * 4);
  float* drv       = (float*)alloc((size_t)NN * 4);
  float* Z         = (float*)alloc((size_t)NN * HH * 4);
  unsigned short* btHi = (unsigned short*)alloc((size_t)NN * HH * 2);
  unsigned short* btLo = (unsigned short*)alloc((size_t)NN * HH * 2);
  float* H0     = (float*)alloc((size_t)NN * HH * 4);
  float* H1     = (float*)alloc((size_t)NN * HH * 4);
  float* pooled = (float*)alloc((size_t)GG * HH * 4);
  int*   offs   = (int*)alloc((GG + 1) * 4);

  k_rowsum<<<NN, 256, 0, stream>>>(adj, d1p, diags);
  k_scales<<<NN / 256, 256, 0, stream>>>(diags, pp, qp, rp, c1p, c2p, c3p,
                                         rowscale, diagscale, drv);

  const float* Hin = x;
  float* Houts[3] = {H0, H1, H0};
  for (int L = 0; L < 3; ++L) {
    k_gemm<<<(NN * HH) / 256, 256, 0, stream>>>(Hin, Ws[L], Z);
    k_split_t<<<dim3(NN / 32, HH / 32), dim3(32, 8), 0, stream>>>(Z, drv, btHi, btLo);
    k_spmm<<<NN / 64, 128, 0, stream>>>(adj, btHi, btLo, Z, drv, rowscale,
                                        diagscale, d1p, bs[L], Houts[L]);
    Hin = Houts[L];
  }

  k_offsets<<<1, 256, 0, stream>>>(batch, offs);
  k_pool<<<GG, HH, 0, stream>>>(Houts[2], offs, pooled);
  k_head<<<1, GG, 0, stream>>>(pooled, fcw, fcb, out);
}
